// selfAttention_24996709662898
// MI455X (gfx1250) — matrix operand order repaired and checked by reading, not compiled, not run on hardware
//
#include <hip/hip_runtime.h>

typedef __attribute__((ext_vector_type(16))) _Float16 v16h;
typedef __attribute__((ext_vector_type(8)))  float    v8f;

#define N_B 2
#define SEQ 2048
#define EMB 1024
#define NH  16
#define HD  64

// ---------------------------------------------------------------------------
// WMMA helper: D = A(16x32 f16) * B(32x16 f16) + C(16x16 f32)
// ---------------------------------------------------------------------------
static __device__ __forceinline__ v8f wmma16(v16h a, v16h b, v8f c) {
  return __builtin_amdgcn_wmma_f32_16x16x32_f16(
      /*neg_a=*/false, a, /*neg_b=*/false, b,
      /*c_mod=*/(short)0, c, /*reuse_a=*/false, /*reuse_b=*/false);
}

// A operand (16x32, f16): lane m = lane&15, g = lane>>4.
// slot h -> K = g*16 + h (contiguous 16 halfs)
static __device__ __forceinline__ v16h load_A_f16(const _Float16* src, int ld) {
  const int lane = threadIdx.x & 31;
  const int m = lane & 15, g = lane >> 4;
  const _Float16* row = src + (size_t)m * ld;
  union { uint4 u[2]; v16h v; } r;
  r.u[0] = *(const uint4*)(row + g * 16);
  r.u[1] = *(const uint4*)(row + g * 16 + 8);
  return r.v;
}

static __device__ __forceinline__ v16h load_A_f32cvt(const float* src, int ld) {
  const int lane = threadIdx.x & 31;
  const int m = lane & 15, g = lane >> 4;
  const float* row = src + (size_t)m * ld;
  v16h a;
#pragma unroll
  for (int h = 0; h < 16; ++h) {
    const int k = g * 16 + h;
    a[h] = (_Float16)row[k];
  }
  return a;
}

// B operand (32x16): lane col n = lane&15, g = lane>>4; slot h -> K = g*16+h.
// Source stored as (col, K) row-major (B transposed): contiguous per lane.
static __device__ __forceinline__ v16h load_B_cols(const _Float16* src, int ld) {
  const int lane = threadIdx.x & 31;
  const int n = lane & 15, g = lane >> 4;
  const _Float16* p = src + (size_t)n * ld + g * 16;
  union { uint4 u[2]; v16h v; } r;
  r.u[0] = ((const uint4*)p)[0];
  r.u[1] = ((const uint4*)p)[1];
  return r.v;
}

static __device__ __forceinline__ v16h load_B_cols_f32cvt(const float* src, int ld) {
  const int lane = threadIdx.x & 31;
  const int n = lane & 15, g = lane >> 4;
  const float* p = src + (size_t)n * ld + g * 16;
  v16h b;
#pragma unroll
  for (int h = 0; h < 16; ++h) b[h] = (_Float16)p[h];
  return b;
}

// ---------------------------------------------------------------------------
// Kernel 1: per-head projection  out = X[n,l,h*D:+D] @ W^T (+ bias)
// transpose_out == 0 : out layout (n, h, l, d)
// transpose_out == 1 : out layout (n, h, d, l)   (used for V; feeds B-operand)
// ---------------------------------------------------------------------------
__global__ void __launch_bounds__(128)
proj_kernel(const float* __restrict__ X, const float* __restrict__ W,
            const float* __restrict__ bias, _Float16* __restrict__ out,
            int has_bias, int transpose_out)
{
  const int wave = threadIdx.x >> 5, lane = threadIdx.x & 31;
  const int wid = blockIdx.x * 4 + wave;
  const int LT = SEQ / 16;
  const int lt = wid % LT;
  const int h  = (wid / LT) % NH;
  const int n  = wid / (LT * NH);
  const int ncol = lane & 15, g = lane >> 4;

  const float* Xrow = X + ((size_t)n * SEQ + (size_t)lt * 16) * EMB + h * HD;
  const v16h a0 = load_A_f32cvt(Xrow, EMB);        // d = 0..31
  const v16h a1 = load_A_f32cvt(Xrow + 32, EMB);   // d = 32..63

  const size_t head = (size_t)(n * NH + h);
#pragma unroll
  for (int eg = 0; eg < 4; ++eg) {
    v8f c = {};
    // B[k][e] = W[e][k]  (torch Linear: x @ W^T)
    c = wmma16(a0, load_B_cols_f32cvt(W + (size_t)(eg * 16) * HD, HD), c);
    c = wmma16(a1, load_B_cols_f32cvt(W + (size_t)(eg * 16) * HD + 32, HD), c);
    const float bv = has_bias ? bias[eg * 16 + ncol] : 0.0f;
    if (transpose_out) {
      // (n,h,d,l): this lane's 8 rows are 8 consecutive l -> one b128 store
      _Float16* o = out + (head * HD + eg * 16 + ncol) * SEQ + (size_t)lt * 16;
      union { uint4 u; _Float16 hlf[8]; } pk;
#pragma unroll
      for (int r = 0; r < 8; ++r) pk.hlf[r] = (_Float16)(c[r] + bv);
      *(uint4*)(o + 8 * g) = pk.u;
    } else {
      _Float16* o = out + (head * SEQ + (size_t)lt * 16) * HD + eg * 16 + ncol;
#pragma unroll
      for (int r = 0; r < 8; ++r)
        o[(size_t)(r + 8 * g) * HD] = (_Float16)(c[r] + bv);
    }
  }
}

// ---------------------------------------------------------------------------
// Kernel 2: flash attention. Block = 4 waves, one (n,h), 64 query rows.
// Each wave: 16 q-rows, loops over 32-wide K/V tiles staged in LDS.
// K tile staged row-major (kv,d) -> Q@K^T B-operand is contiguous.
// V tile staged transposed (d,kv) -> P@V  B-operand is contiguous.
// ---------------------------------------------------------------------------
__global__ void __launch_bounds__(128)
attn_kernel(const _Float16* __restrict__ Qp, const _Float16* __restrict__ Kp,
            const _Float16* __restrict__ Vtp, const int* __restrict__ mask,
            _Float16* __restrict__ Oattn)
{
  __shared__ __align__(16) _Float16 sK [32 * HD];       // 4 KB (kv, d)
  __shared__ __align__(16) _Float16 sVt[HD * 32];       // 4 KB (d, kv)
  __shared__ __align__(16) _Float16 sP [4 * 16 * 32];   // 4 KB per-wave P tiles

  const int wave = threadIdx.x >> 5, lane = threadIdx.x & 31;
  const int ncol = lane & 15, g = lane >> 4;
  const int qt = (blockIdx.x & 31) * 4 + wave;          // 0..127
  const int h  = (blockIdx.x >> 5) & 15;
  const int n  = blockIdx.x >> 9;
  const int q0 = qt * 16;

  const size_t headBase = ((size_t)n * NH + h) * SEQ * HD;  // same for (l,d) and (d,l)
  const _Float16* Qt = Qp + headBase + (size_t)q0 * HD;
  const v16h aq0 = load_A_f16(Qt, HD);
  const v16h aq1 = load_A_f16(Qt + 32, HD);

  v8f o0 = {}, o1 = {}, o2 = {}, o3 = {};
  float row_m[8], row_l[8];
#pragma unroll
  for (int r = 0; r < 8; ++r) { row_m[r] = -3.0e38f; row_l[r] = 0.0f; }

  const float inv_sqrt_E = 0.03125f;                    // 1/sqrt(1024)
  _Float16* Pw = sP + wave * (16 * 32);

  for (int kv0 = 0; kv0 < SEQ; kv0 += 32) {
    const _Float16* Kt  = Kp  + headBase + (size_t)kv0 * HD;  // (kv, d) tile
    const _Float16* Vtt = Vtp + headBase + kv0;               // (d, kv) tile

    __syncthreads();  // previous tile fully consumed
    {
      const int t = threadIdx.x;
      // K tile: 32x64 f16 contiguous 4 KB: 256 uint4, 2 per thread
      ((uint4*)sK)[t]       = ((const uint4*)Kt)[t];
      ((uint4*)sK)[t + 128] = ((const uint4*)Kt)[t + 128];
      // V^T tile: 64 rows (d) x 32 halfs, row stride SEQ: 256 uint4, 2/thread
#pragma unroll
      for (int i = 0; i < 2; ++i) {
        const int c2 = t + i * 128;           // uint4 index: d*4 + part
        const int d = c2 >> 2, part = c2 & 3;
        ((uint4*)sVt)[c2] = *(const uint4*)(Vtt + (size_t)d * SEQ + part * 8);
      }
      if (kv0 + 32 < SEQ) {
        __builtin_prefetch((const void*)(Kt + 32 * HD), 0, 3);
        __builtin_prefetch((const void*)(Vtt + 32), 0, 3);
      }
    }
    __syncthreads();

    // S = Q @ K^T : two 16x16 column groups, K-dim = 64 (two WMMA chunks)
    v8f s0 = {}, s1 = {};
    s0 = wmma16(aq0, load_B_cols(sK, HD), s0);
    s0 = wmma16(aq1, load_B_cols(sK + 32, HD), s0);
    s1 = wmma16(aq0, load_B_cols(sK + 16 * HD, HD), s1);
    s1 = wmma16(aq1, load_B_cols(sK + 16 * HD + 32, HD), s1);

    // mask (per key column, broadcast over heads/queries) then scale
    const int mv0 = mask[(size_t)n * SEQ + kv0 + ncol];
    const int mv1 = mask[(size_t)n * SEQ + kv0 + 16 + ncol];
    float f0[8], f1[8];
#pragma unroll
    for (int r = 0; r < 8; ++r) {
      f0[r] = (mv0 ? (float)s0[r] : -1.0e20f) * inv_sqrt_E;
      f1[r] = (mv1 ? (float)s1[r] : -1.0e20f) * inv_sqrt_E;
    }

    // online softmax: row max + rescale (rows live across 16-lane half-groups)
    float mnew[8], al[8];
#pragma unroll
    for (int r = 0; r < 8; ++r) {
      float m = fmaxf(f0[r], f1[r]);
#pragma unroll
      for (int off = 1; off < 16; off <<= 1) m = fmaxf(m, __shfl_xor(m, off, 32));
      mnew[r] = fmaxf(row_m[r], m);
      al[r] = __expf(row_m[r] - mnew[r]);
      row_m[r] = mnew[r];
    }

    float p0[8], p1[8];
#pragma unroll
    for (int r = 0; r < 8; ++r) {
      p0[r] = __expf(f0[r] - mnew[r]);
      p1[r] = __expf(f1[r] - mnew[r]);
      float rs = p0[r] + p1[r];
#pragma unroll
      for (int off = 1; off < 16; off <<= 1) rs += __shfl_xor(rs, off, 32);
      row_l[r] = row_l[r] * al[r] + rs;
      o0[r] *= al[r]; o1[r] *= al[r]; o2[r] *= al[r]; o3[r] *= al[r];
    }

    // C-layout -> A-layout reorg through LDS (row-major 16x32 f16 per wave)
#pragma unroll
    for (int r = 0; r < 8; ++r) {
      const int M = r + 8 * g;
      Pw[M * 32 + ncol]      = (_Float16)p0[r];
      Pw[M * 32 + 16 + ncol] = (_Float16)p1[r];
    }
    __syncthreads();

    // O += P(16x32) @ V(32x64): one A frag, 4 column groups from V^T tile
    const v16h ap = load_A_f16(Pw, 32);
    o0 = wmma16(ap, load_B_cols(sVt +  0 * 32, 32), o0);
    o1 = wmma16(ap, load_B_cols(sVt + 16 * 32, 32), o1);
    o2 = wmma16(ap, load_B_cols(sVt + 32 * 32, 32), o2);
    o3 = wmma16(ap, load_B_cols(sVt + 48 * 32, 32), o3);
  }

  // epilogue: normalize and store f16 in (n, l, h, d) = row-major (n*l, e)
  float inv[8];
#pragma unroll
  for (int r = 0; r < 8; ++r) inv[r] = (row_l[r] > 0.0f) ? (1.0f / row_l[r]) : 0.0f;

  _Float16* ob = Oattn + (((size_t)n * SEQ + q0) * NH + h) * HD;
#pragma unroll
  for (int r = 0; r < 8; ++r) {
    const size_t rowoff = (size_t)(r + 8 * g) * EMB;   // H*D == EMB stride
    ob[rowoff +  0 + ncol] = (_Float16)(o0[r] * inv[r]);
    ob[rowoff + 16 + ncol] = (_Float16)(o1[r] * inv[r]);
    ob[rowoff + 32 + ncol] = (_Float16)(o2[r] * inv[r]);
    ob[rowoff + 48 + ncol] = (_Float16)(o3[r] * inv[r]);
  }
}

// ---------------------------------------------------------------------------
// Kernel 3: out = Attn(4096 x 1024, f16) @ Wo^T + bo  -> f32
// Block = 4 waves covering 64 rows x 16 cols; Wo tile staged transposed
// ([n][k], f16) in LDS so the B-operand load is two b128 per lane.
// ---------------------------------------------------------------------------
__global__ void __launch_bounds__(128)
out_proj_kernel(const _Float16* __restrict__ A, const float* __restrict__ Wo,
                const float* __restrict__ bo, float* __restrict__ out)
{
  __shared__ __align__(16) _Float16 sB[16 * 32];       // B^T tile [n][k], 1 KB

  const int wave = threadIdx.x >> 5, lane = threadIdx.x & 31;
  const int ncol = lane & 15, g = lane >> 4;
  const int ct = blockIdx.x & 63;                      // column tile 0..63
  const int rg = blockIdx.x >> 6;                      // row group 0..63
  const int row0 = (rg * 4 + wave) * 16;
  const int col0 = ct * 16;

  v8f c = {};
  for (int kc = 0; kc < EMB / 32; ++kc) {
    __syncthreads();
    // stage sB[n][k] = Wo[col0+n][kc*32+k] as f16: 512 elems, 4 contiguous
    // k per thread -> coalesced f32 reads, mergeable f16 LDS writes
#pragma unroll
    for (int i = 0; i < 4; ++i) {
      const int idx = threadIdx.x * 4 + i;
      const int nn = idx >> 5, kk = idx & 31;
      sB[nn * 32 + kk] =
          (_Float16)Wo[(size_t)(col0 + nn) * EMB + kc * 32 + kk];
    }
    __syncthreads();
    const v16h a = load_A_f16(A + (size_t)row0 * EMB + kc * 32, EMB);
    c = wmma16(a, load_B_cols(sB, 32), c);
  }

  const float bias = bo[col0 + ncol];
#pragma unroll
  for (int r = 0; r < 8; ++r)
    out[(size_t)(row0 + r + 8 * g) * EMB + col0 + ncol] = c[r] + bias;
}

// ---------------------------------------------------------------------------
// Host launcher
// ---------------------------------------------------------------------------
extern "C" void kernel_launch(void* const* d_in, const int* in_sizes, int n_in,
                              void* d_out, int out_size, void* d_ws, size_t ws_size,
                              hipStream_t stream) {
  (void)in_sizes; (void)n_in; (void)out_size; (void)ws_size;
  const float* values = (const float*)d_in[0];
  const float* key    = (const float*)d_in[1];
  const float* query  = (const float*)d_in[2];
  const int*   mask   = (const int*)d_in[3];
  const float* Wv     = (const float*)d_in[4];
  const float* Wk     = (const float*)d_in[5];
  const float* Wq     = (const float*)d_in[6];
  const float* bq     = (const float*)d_in[7];
  const float* Wo     = (const float*)d_in[8];
  const float* bo     = (const float*)d_in[9];
  float* out = (float*)d_out;

  const size_t per = (size_t)N_B * NH * SEQ * HD;      // 4M halfs each
  _Float16* Vt = (_Float16*)d_ws;                      // (n,h,d,l) transposed
  _Float16* Kp = Vt + per;                             // (n,h,l,d)
  _Float16* Qp = Kp + per;                             // (n,h,l,d)
  _Float16* Oa = Qp + per;                             // (N*L, E) f16

  const dim3 blk(128);
  proj_kernel<<<1024, blk, 0, stream>>>(values, Wv, bq, Vt, 0, 1);
  proj_kernel<<<1024, blk, 0, stream>>>(key,    Wk, bq, Kp, 0, 0);
  proj_kernel<<<1024, blk, 0, stream>>>(query,  Wq, bq, Qp, 1, 0);
  attn_kernel<<<1024, blk, 0, stream>>>(Qp, Kp, Vt, mask, Oa);
  out_proj_kernel<<<4096, blk, 0, stream>>>(Oa, Wo, bo, out);
}
